// GQAAttention_55473797595670
// MI455X (gfx1250) — matrix operand order repaired and checked by reading, not compiled, not run on hardware
//
#include <hip/hip_runtime.h>
#include <stdint.h>

#define HIDDEN 2048
#define NUM_HEADS 16
#define NUM_KV_HEADS 4
#define HEAD_DIM 128
#define SEQ 4096

typedef __attribute__((ext_vector_type(16))) __bf16 v16bf;
typedef __attribute__((ext_vector_type(8)))  float  v8f;
typedef __attribute__((ext_vector_type(4)))  unsigned int v4u;
typedef __attribute__((ext_vector_type(4)))  unsigned int u32x4;
typedef __attribute__((ext_vector_type(8)))  int i32x8;
typedef __attribute__((ext_vector_type(4)))  int i32x4;

__device__ __forceinline__ unsigned short f32_to_bf16(float f) {
  unsigned int u = __float_as_uint(f);
  u += 0x7FFFu + ((u >> 16) & 1u);  // round-to-nearest-even
  return (unsigned short)(u >> 16);
}
__device__ __forceinline__ float bf16_to_f32(unsigned short h) {
  return __uint_as_float(((unsigned int)h) << 16);
}

union AFrag {
  v16bf v;
  v4u   q[2];
};

// ---------------------------------------------------------------------------
// Tensor Data Mover: 2D bf16 tile load global -> LDS with optional row pad.
// D# layout per CDNA5 ISA ch.8 (CDNA5 ISA):
//   group0: {flags(count=1), lds_addr, gaddr[31:0], gaddr[56:32]|type=2<<30}
//   group1: dw0 = data_size=1(2B) | pad_enable | pad_interval<<22 | pad_amount<<25
//           dw1[31:16]=tensor_dim0[15:0]   dw2 = dim0[31:16] | dim1[15:0]<<16
//           dw3 = dim1[31:16] | tile_dim0<<16   dw4 = tile_dim1 (tile_dim2=0)
//           dw5 = tensor_dim0_stride[31:0], dw6/dw7 = 0
//   groups 2/3: zero (2-D tile)
// pad_interval encodes 2^(v+1) dwords between pads; pad_amount = dwords-1.
// ---------------------------------------------------------------------------
__device__ __forceinline__ void tdm_load_2d(void* lds_dst,
                                            const unsigned short* gsrc,
                                            unsigned tile_w, unsigned tile_h,
                                            unsigned stride_elems,
                                            unsigned pad_interval,
                                            unsigned pad_amount) {
  unsigned lds_off = (unsigned)(uintptr_t)lds_dst;  // low 32 bits = LDS offset
  unsigned long long ga = (unsigned long long)(uintptr_t)gsrc;
  u32x4 g0;
  g0[0] = 1u;  // count=1, user descriptor
  g0[1] = lds_off;
  g0[2] = (unsigned)ga;
  g0[3] = ((unsigned)(ga >> 32) & 0x01ffffffu) | (2u << 30);  // type=2 "image"
  i32x8 g1;
  g1[0] = (int)((1u << 16) |                 // data_size = 2 bytes
                (1u << 20) |                 // pad_enable
                (pad_interval << 22) | (pad_amount << 25));
  g1[1] = (int)((tile_w & 0xffffu) << 16);   // tensor_dim0 lo (== tile_w)
  g1[2] = (int)(((tile_w >> 16) & 0xffffu) | ((tile_h & 0xffffu) << 16));
  g1[3] = (int)(((tile_h >> 16) & 0xffffu) | ((tile_w & 0xffffu) << 16));
  g1[4] = (int)(tile_h & 0xffffu);           // tile_dim1, tile_dim2=0
  g1[5] = (int)stride_elems;                 // tensor_dim0_stride lo32
  g1[6] = 0;
  g1[7] = 0;
  i32x4 z4 = {0, 0, 0, 0};
#if __clang_major__ >= 23
  i32x8 z8 = {0, 0, 0, 0, 0, 0, 0, 0};
  __builtin_amdgcn_tensor_load_to_lds(g0, g1, z4, z4, z8, 0);
#else
  __builtin_amdgcn_tensor_load_to_lds(g0, g1, z4, z4, 0);
#endif
}

// ---------------------------------------------------------------------------
// f32 -> bf16 conversion
// ---------------------------------------------------------------------------
__global__ void cvt_f32_bf16(const float* __restrict__ in,
                             unsigned short* __restrict__ out, int n) {
  int i = blockIdx.x * blockDim.x + threadIdx.x;
  if (i < n) out[i] = f32_to_bf16(in[i]);
}

// ---------------------------------------------------------------------------
// GEMM: C[M][N] = A[M][K] * B[N][K]^T  (A,B bf16 row-major, K-contiguous).
// 256 threads = 8 waves, block tile 128x128, K-step 32.
// TDM double-buffer: wave0 issues tensor_load_to_lds for tile i+1, all waves
// run WMMA on tile i from LDS, wave0 s_wait_tensorcnt, one barrier per step.
// LDS rows: 32 elems (16 dwords) + 4 dword pad -> pad_interval=3, amount=3.
// ---------------------------------------------------------------------------
template <int OUT_BF16>
__global__ __launch_bounds__(256) void gemm_bf16_nt(
    const unsigned short* __restrict__ A, const unsigned short* __restrict__ B,
    void* __restrict__ Cout, int M, int N, int K) {
  __shared__ alignas(16) unsigned short As[2][128][40];
  __shared__ alignas(16) unsigned short Bs[2][128][40];

  const int m0 = blockIdx.y * 128;
  const int n0 = blockIdx.x * 128;
  const int tid = threadIdx.x;
  const int lane = tid & 31, wave = tid >> 5;
  const int half = lane >> 4, l16 = lane & 15;
  const int wm = (wave & 3) * 32;
  const int wn = (wave >> 2) * 64;
  const bool issuer = (wave == 0);

  v8f acc[2][4];
#pragma unroll
  for (int i = 0; i < 2; i++)
#pragma unroll
    for (int j = 0; j < 4; j++)
#pragma unroll
      for (int e = 0; e < 8; e++) acc[i][j][e] = 0.0f;

  if (issuer) {
    tdm_load_2d(&As[0][0][0], &A[(size_t)m0 * K], 32, 128, (unsigned)K, 3, 3);
    tdm_load_2d(&Bs[0][0][0], &B[(size_t)n0 * K], 32, 128, (unsigned)K, 3, 3);
    __builtin_amdgcn_s_wait_tensorcnt(0);
  }
  __syncthreads();

  int buf = 0;
  for (int kk = 0; kk < K; kk += 32) {
    const bool have_next = (kk + 32) < K;
    if (have_next && issuer) {
      tdm_load_2d(&As[buf ^ 1][0][0], &A[(size_t)m0 * K + kk + 32], 32, 128,
                  (unsigned)K, 3, 3);
      tdm_load_2d(&Bs[buf ^ 1][0][0], &B[(size_t)n0 * K + kk + 32], 32, 128,
                  (unsigned)K, 3, 3);
    }

    AFrag af[2], bf_[4];
#pragma unroll
    for (int i = 0; i < 2; i++) {
      int r = wm + i * 16 + l16;
      int kb = half * 8;  // A frag: lanes0-15 K{0-7,16-23}, lanes16-31 +8
      af[i].q[0] = *reinterpret_cast<const v4u*>(&As[buf][r][kb]);
      af[i].q[1] = *reinterpret_cast<const v4u*>(&As[buf][r][kb + 16]);
    }
#pragma unroll
    for (int j = 0; j < 4; j++) {
      int r = wn + j * 16 + l16;
      int kb = half * 8;  // B frag: lanes0-15 K{0-7,16-23}, lanes16-31 +8
      bf_[j].q[0] = *reinterpret_cast<const v4u*>(&Bs[buf][r][kb]);
      bf_[j].q[1] = *reinterpret_cast<const v4u*>(&Bs[buf][r][kb + 16]);
    }
#pragma unroll
    for (int i = 0; i < 2; i++)
#pragma unroll
      for (int j = 0; j < 4; j++)
        acc[i][j] = __builtin_amdgcn_wmma_f32_16x16x32_bf16(
            false, af[i].v, false, bf_[j].v, (short)0, acc[i][j], false, false);

    if (have_next && issuer) __builtin_amdgcn_s_wait_tensorcnt(0);
    __syncthreads();
    buf ^= 1;
  }

#pragma unroll
  for (int i = 0; i < 2; i++)
#pragma unroll
    for (int j = 0; j < 4; j++)
#pragma unroll
      for (int r = 0; r < 8; r++) {
        int gm = m0 + wm + i * 16 + r + half * 8;  // C: M = r + 8*(lane/16)
        int gn = n0 + wn + j * 16 + l16;           //    N = lane%16
        float val = acc[i][j][r];
        if (OUT_BF16)
          ((unsigned short*)Cout)[(size_t)gm * N + gn] = f32_to_bf16(val);
        else
          ((float*)Cout)[(size_t)gm * N + gn] = val;
      }
}

// ---------------------------------------------------------------------------
// In-place RoPE on bf16 [SEQ][nheads*HEAD_DIM] (interleaved pairs)
// ---------------------------------------------------------------------------
__global__ void rope_inplace(unsigned short* __restrict__ buf, int nheads) {
  int idx = blockIdx.x * blockDim.x + threadIdx.x;
  int total = SEQ * nheads * (HEAD_DIM / 2);
  if (idx >= total) return;
  int i = idx % (HEAD_DIM / 2);
  int rem = idx / (HEAD_DIM / 2);
  int hh = rem % nheads;
  int s = rem / nheads;
  size_t base = (size_t)s * nheads * HEAD_DIM + (size_t)hh * HEAD_DIM + 2 * i;
  float x1 = bf16_to_f32(buf[base]);
  float x2 = bf16_to_f32(buf[base + 1]);
  float freq = __expf(-((float)(2 * i) / (float)HEAD_DIM) * 9.210340371976184f);
  float ang = (float)s * freq;
  float sn, c;
  __sincosf(ang, &sn, &c);
  buf[base] = f32_to_bf16(x1 * c - x2 * sn);
  buf[base + 1] = f32_to_bf16(x1 * sn + x2 * c);
}

// ---------------------------------------------------------------------------
// Flash attention, causal GQA. 128 threads = 4 waves, Br=Bc=64.
// Q and K tiles arrive via TDM (wave0 issues; V load+transpose overlaps on the
// vector pipe). Each wave owns 16 q-rows end-to-end: 4 score WMMA tiles,
// online softmax via shfl_xor in 16-lane halves, P->LDS->A-frag, 16 PV WMMAs.
// LDS rows 128 elems (64 dwords) + 4 dword pad -> pad_interval=5, amount=3.
// ---------------------------------------------------------------------------
__global__ __launch_bounds__(128) void flash_attn(
    const unsigned short* __restrict__ Qb,  // [SEQ][16*128] bf16 (post-RoPE)
    const unsigned short* __restrict__ Kb,  // [SEQ][4*128]  bf16 (post-RoPE)
    const unsigned short* __restrict__ Vb,  // [SEQ][4*128]  bf16
    unsigned short* __restrict__ Ob) {      // [SEQ][16*128] bf16
  __shared__ alignas(16) unsigned short Qs[64][136];
  __shared__ alignas(16) unsigned short Ks[64][136];
  __shared__ alignas(16) unsigned short Vt[128][72];    // transposed [dim][key]
  __shared__ alignas(16) unsigned short Ps[4][16][72];  // per-wave P tile

  const int qb = blockIdx.x;  // 0..63
  const int h = blockIdx.y;   // 0..15
  const int kvh = h >> 2;
  const int tid = threadIdx.x;
  const int lane = tid & 31, wave = tid >> 5;
  const int half = lane >> 4, l16 = lane & 15;
  const int q0 = qb * 64;
  const float scale = 0.08838834764831845f;  // 1/sqrt(128)

  if (wave == 0) {  // async Q tile: waited before first compute barrier
    tdm_load_2d(&Qs[0][0], &Qb[(size_t)q0 * (NUM_HEADS * HEAD_DIM) +
                               h * HEAD_DIM],
                128, 64, NUM_HEADS * HEAD_DIM, 5, 3);
  }

  v8f o_acc[8];
#pragma unroll
  for (int nt = 0; nt < 8; nt++)
#pragma unroll
    for (int e = 0; e < 8; e++) o_acc[nt][e] = 0.0f;
  float m_run[8], l_run[8];
#pragma unroll
  for (int r = 0; r < 8; r++) {
    m_run[r] = -1e30f;
    l_run[r] = 0.0f;
  }

  for (int kb_i = 0; kb_i <= qb; kb_i++) {
    const int k0 = kb_i * 64;
    __syncthreads();  // previous iteration done reading Ks/Vt
    if (wave == 0) {
      tdm_load_2d(&Ks[0][0], &Kb[(size_t)k0 * (NUM_KV_HEADS * HEAD_DIM) +
                                 kvh * HEAD_DIM],
                  128, 64, NUM_KV_HEADS * HEAD_DIM, 5, 3);
    }
    // V tile load + transpose into Vt (overlaps the TDM)
#pragma unroll
    for (int c = 0; c < 8; c++) {
      int id = tid * 8 + c;
      int row = id >> 4;
      int col = (id & 15) * 8;
      v4u vv = *reinterpret_cast<const v4u*>(
          &Vb[(size_t)(k0 + row) * (NUM_KV_HEADS * HEAD_DIM) + kvh * HEAD_DIM +
              col]);
#pragma unroll
      for (int i2 = 0; i2 < 4; i2++) {
        unsigned int w = vv[i2];
        Vt[col + i2 * 2 + 0][row] = (unsigned short)(w & 0xffffu);
        Vt[col + i2 * 2 + 1][row] = (unsigned short)(w >> 16);
      }
    }
    if (wave == 0) __builtin_amdgcn_s_wait_tensorcnt(0);
    __syncthreads();

    // ---- scores: S_w(16x64) = Q_w(16x128) @ K_tile^T ----
    v8f s_acc[4];
#pragma unroll
    for (int t = 0; t < 4; t++)
#pragma unroll
      for (int e = 0; e < 8; e++) s_acc[t][e] = 0.0f;
#pragma unroll
    for (int ks = 0; ks < 4; ks++) {
      AFrag aq;
      int qr = wave * 16 + l16;
      int kb2 = ks * 32 + half * 8;
      aq.q[0] = *reinterpret_cast<const v4u*>(&Qs[qr][kb2]);
      aq.q[1] = *reinterpret_cast<const v4u*>(&Qs[qr][kb2 + 16]);
#pragma unroll
      for (int t = 0; t < 4; t++) {
        AFrag bk;
        int kr = t * 16 + l16;
        int kb3 = ks * 32 + half * 8;
        bk.q[0] = *reinterpret_cast<const v4u*>(&Ks[kr][kb3]);
        bk.q[1] = *reinterpret_cast<const v4u*>(&Ks[kr][kb3 + 16]);
        s_acc[t] = __builtin_amdgcn_wmma_f32_16x16x32_bf16(
            false, aq.v, false, bk.v, (short)0, s_acc[t], false, false);
      }
    }

    // ---- scale + causal mask + row max ----
    const bool diag = (kb_i == qb);
    float rmax[8];
#pragma unroll
    for (int r = 0; r < 8; r++) rmax[r] = -1e30f;
#pragma unroll
    for (int t = 0; t < 4; t++)
#pragma unroll
      for (int r = 0; r < 8; r++) {
        float sv = s_acc[t][r] * scale;
        if (diag) {
          int mg = wave * 16 + r + half * 8;
          int ng = t * 16 + l16;
          if (ng > mg) sv = -1e30f;
        }
        s_acc[t][r] = sv;
        rmax[r] = fmaxf(rmax[r], sv);
      }
#pragma unroll
    for (int r = 0; r < 8; r++) {
#pragma unroll
      for (int msk = 1; msk < 16; msk <<= 1)
        rmax[r] = fmaxf(rmax[r], __shfl_xor(rmax[r], msk, 32));
    }

    // ---- online softmax update ----
    float alpha[8], lsum[8];
#pragma unroll
    for (int r = 0; r < 8; r++) {
      float mn = fmaxf(m_run[r], rmax[r]);
      alpha[r] = __expf(m_run[r] - mn);
      m_run[r] = mn;
      lsum[r] = 0.0f;
    }
#pragma unroll
    for (int t = 0; t < 4; t++)
#pragma unroll
      for (int r = 0; r < 8; r++) {
        float pv = __expf(s_acc[t][r] - m_run[r]);
        lsum[r] += pv;
        Ps[wave][r + half * 8][t * 16 + l16] = f32_to_bf16(pv);
      }
#pragma unroll
    for (int r = 0; r < 8; r++) {
#pragma unroll
      for (int msk = 1; msk < 16; msk <<= 1)
        lsum[r] += __shfl_xor(lsum[r], msk, 32);
      l_run[r] = l_run[r] * alpha[r] + lsum[r];
    }
#pragma unroll
    for (int nt = 0; nt < 8; nt++)
#pragma unroll
      for (int r = 0; r < 8; r++) o_acc[nt][r] *= alpha[r];

    // ---- O += P(16x64) @ V(64x128), P via per-wave LDS relayout ----
#pragma unroll
    for (int ks2 = 0; ks2 < 2; ks2++) {
      AFrag ap;
      int kb4 = ks2 * 32 + half * 8;
      ap.q[0] = *reinterpret_cast<const v4u*>(&Ps[wave][l16][kb4]);
      ap.q[1] = *reinterpret_cast<const v4u*>(&Ps[wave][l16][kb4 + 16]);
#pragma unroll
      for (int nt = 0; nt < 8; nt++) {
        AFrag bv;
        int vr = nt * 16 + l16;
        int kb5 = ks2 * 32 + half * 8;
        bv.q[0] = *reinterpret_cast<const v4u*>(&Vt[vr][kb5]);
        bv.q[1] = *reinterpret_cast<const v4u*>(&Vt[vr][kb5 + 16]);
        o_acc[nt] = __builtin_amdgcn_wmma_f32_16x16x32_bf16(
            false, ap.v, false, bv.v, (short)0, o_acc[nt], false, false);
      }
    }
  }

  // ---- epilogue: O /= l, store bf16 ----
#pragma unroll
  for (int nt = 0; nt < 8; nt++)
#pragma unroll
    for (int r = 0; r < 8; r++) {
      int gm = q0 + wave * 16 + r + half * 8;
      int gn = h * HEAD_DIM + nt * 16 + l16;
      float ov = o_acc[nt][r] / l_run[r];
      Ob[(size_t)gm * (NUM_HEADS * HEAD_DIM) + gn] = f32_to_bf16(ov);
    }
}

// ---------------------------------------------------------------------------
// Launcher
// ---------------------------------------------------------------------------
extern "C" void kernel_launch(void* const* d_in, const int* in_sizes, int n_in,
                              void* d_out, int out_size, void* d_ws,
                              size_t ws_size, hipStream_t stream) {
  (void)in_sizes; (void)n_in; (void)out_size; (void)ws_size;
  const float* x = (const float*)d_in[0];
  const float* Wq = (const float*)d_in[1];
  const float* Wk = (const float*)d_in[2];
  const float* Wv = (const float*)d_in[3];
  const float* Wo = (const float*)d_in[4];
  float* out = (float*)d_out;

  char* p = (char*)d_ws;
  auto alloc = [&](size_t bytes) -> char* {
    char* r = p;
    p += (bytes + 255) & ~(size_t)255;
    return r;
  };
  unsigned short* xb = (unsigned short*)alloc((size_t)SEQ * HIDDEN * 2);
  unsigned short* wqb =
      (unsigned short*)alloc((size_t)NUM_HEADS * HEAD_DIM * HIDDEN * 2);
  unsigned short* wkb =
      (unsigned short*)alloc((size_t)NUM_KV_HEADS * HEAD_DIM * HIDDEN * 2);
  unsigned short* wvb =
      (unsigned short*)alloc((size_t)NUM_KV_HEADS * HEAD_DIM * HIDDEN * 2);
  unsigned short* wob =
      (unsigned short*)alloc((size_t)HIDDEN * NUM_HEADS * HEAD_DIM * 2);
  unsigned short* qbuf =
      (unsigned short*)alloc((size_t)SEQ * NUM_HEADS * HEAD_DIM * 2);
  unsigned short* kbuf =
      (unsigned short*)alloc((size_t)SEQ * NUM_KV_HEADS * HEAD_DIM * 2);
  unsigned short* vbuf =
      (unsigned short*)alloc((size_t)SEQ * NUM_KV_HEADS * HEAD_DIM * 2);
  unsigned short* obuf =
      (unsigned short*)alloc((size_t)SEQ * NUM_HEADS * HEAD_DIM * 2);

  auto cvt = [&](const float* src, unsigned short* dst, int n) {
    cvt_f32_bf16<<<(n + 255) / 256, 256, 0, stream>>>(src, dst, n);
  };
  cvt(x, xb, SEQ * HIDDEN);
  cvt(Wq, wqb, NUM_HEADS * HEAD_DIM * HIDDEN);
  cvt(Wk, wkb, NUM_KV_HEADS * HEAD_DIM * HIDDEN);
  cvt(Wv, wvb, NUM_KV_HEADS * HEAD_DIM * HIDDEN);
  cvt(Wo, wob, HIDDEN * NUM_HEADS * HEAD_DIM);

  gemm_bf16_nt<1><<<dim3((NUM_HEADS * HEAD_DIM) / 128, SEQ / 128), 256, 0,
                    stream>>>(xb, wqb, qbuf, SEQ, NUM_HEADS * HEAD_DIM, HIDDEN);
  gemm_bf16_nt<1><<<dim3((NUM_KV_HEADS * HEAD_DIM) / 128, SEQ / 128), 256, 0,
                    stream>>>(xb, wkb, kbuf, SEQ, NUM_KV_HEADS * HEAD_DIM,
                              HIDDEN);
  gemm_bf16_nt<1><<<dim3((NUM_KV_HEADS * HEAD_DIM) / 128, SEQ / 128), 256, 0,
                    stream>>>(xb, wvb, vbuf, SEQ, NUM_KV_HEADS * HEAD_DIM,
                              HIDDEN);

  rope_inplace<<<(SEQ * NUM_HEADS * (HEAD_DIM / 2) + 255) / 256, 256, 0,
                 stream>>>(qbuf, NUM_HEADS);
  rope_inplace<<<(SEQ * NUM_KV_HEADS * (HEAD_DIM / 2) + 255) / 256, 256, 0,
                 stream>>>(kbuf, NUM_KV_HEADS);

  flash_attn<<<dim3(SEQ / 64, NUM_HEADS), 128, 0, stream>>>(qbuf, kbuf, vbuf,
                                                            obuf);

  gemm_bf16_nt<0><<<dim3(HIDDEN / 128, SEQ / 128), 256, 0, stream>>>(
      obuf, wob, out, SEQ, HIDDEN, NUM_HEADS * HEAD_DIM);
}
